// Wide_Conv_22093311771259
// MI455X (gfx1250) — matrix operand order repaired and checked by reading, not compiled, not run on hardware
//
#include <hip/hip_runtime.h>
#include <hip/hip_bf16.h>
#include <math.h>

typedef __attribute__((ext_vector_type(16))) _Float16 v16h;
typedef __attribute__((ext_vector_type(8)))  _Float16 v8h;
typedef __attribute__((ext_vector_type(8)))  float    v8f;

#define B_ 16
#define L_ 128
#define D_ 256

// ---- WMMA fragment loaders (wave32 layouts per CDNA5 ISA 7.12.2) ----------
// A-matrix 16x32 f16: lane = M (lane&15), element e -> K = (e<8 ? 8*half+e
// : 16+8*half+(e-8)).  Two contiguous 8xf16 (16B) loads per lane.
__device__ __forceinline__ v16h load_a_frag(const _Float16* rowk0, int half) {
    v8h lo = *(const v8h*)(rowk0 + 8 * half);
    v8h hi = *(const v8h*)(rowk0 + 16 + 8 * half);
    v16h a;
#pragma unroll
    for (int i = 0; i < 8; ++i) { a[i] = lo[i]; a[i + 8] = hi[i]; }
    return a;
}

// B-matrix 32x16 f16: lane = N (lane&15), element e -> K = (e<8 ? 8*half+e
// : 16+8*half+(e-8)).
// Source must be K-contiguous per N-row (we pre-transpose to guarantee it).
__device__ __forceinline__ v16h load_b_frag(const _Float16* colk0, int half) {
    v8h lo = *(const v8h*)(colk0 + 8 * half);
    v8h hi = *(const v8h*)(colk0 + 16 + 8 * half);
    v16h b;
#pragma unroll
    for (int i = 0; i < 8; ++i) { b[i] = lo[i]; b[i + 8] = hi[i]; }
    return b;
}

// ---- Kernel 1: mask, f32->f16 convert, per-row squared norms --------------
__global__ __launch_bounds__(256) void prep_rows_kernel(
    const float* __restrict__ sent1, const float* __restrict__ sent2,
    const int* __restrict__ mask1, const int* __restrict__ mask2,
    _Float16* __restrict__ s1h, _Float16* __restrict__ s2h,
    float* __restrict__ n1, float* __restrict__ n2) {
    __shared__ float red[256];
    const int row   = blockIdx.x;          // b*L + l
    const int which = blockIdx.y;          // 0 -> sent1, 1 -> sent2
    const float* sent = which ? sent2 : sent1;
    const int*   mask = which ? mask2 : mask1;
    _Float16*    sh   = which ? s2h : s1h;
    float*       nrm  = which ? n2 : n1;
    const int tid = threadIdx.x;
    const float keep = (mask[row] == 0) ? 1.0f : 0.0f;
    const float v = sent[(size_t)row * D_ + tid] * keep;
    sh[(size_t)row * D_ + tid] = (_Float16)v;
    red[tid] = v * v;
    __syncthreads();
#pragma unroll
    for (int s = 128; s > 0; s >>= 1) {
        if (tid < s) red[tid] += red[tid + s];
        __syncthreads();
    }
    if (tid == 0) nrm[row] = red[0];
}

// ---- Kernel 2: W [L,D] f32 -> W^T [D,L] f16 (K-contiguous B operand) ------
__global__ __launch_bounds__(256) void prep_w_kernel(
    const float* __restrict__ W, _Float16* __restrict__ Wht) {
    const int t = blockIdx.x;     // 0..127
    const int d = threadIdx.x;    // 0..255
    Wht[(size_t)d * L_ + t] = (_Float16)W[(size_t)t * D_ + d];
}

// ---- Kernel 3: G = s1 @ s2^T via WMMA, epilogue A = 1/(1+sqrt(d2)) --------
// grid (B, L/16), 256 threads = 8 waves; wave w owns t-tile w.
__global__ __launch_bounds__(256) void dist_kernel(
    const _Float16* __restrict__ s1h, const _Float16* __restrict__ s2h,
    const float* __restrict__ n1, const float* __restrict__ n2,
    _Float16* __restrict__ Ah, _Float16* __restrict__ Aht) {
    const int b    = blockIdx.x;
    const int s0   = blockIdx.y * 16;
    const int wave = threadIdx.x >> 5;
    const int lane = threadIdx.x & 31;
    const int half = lane >> 4;
    const int l15  = lane & 15;
    const int t0   = wave * 16;
    const _Float16* s1row = s1h + (size_t)b * L_ * D_ + (size_t)(s0 + l15) * D_;
    const _Float16* s2row = s2h + (size_t)b * L_ * D_ + (size_t)(t0 + l15) * D_;
    v8f acc = {};
#pragma unroll
    for (int k0 = 0; k0 < D_; k0 += 32) {
        v16h a  = load_a_frag(s1row + k0, half);      // A = s1 (M=s, K=d)
        v16h bb = load_b_frag(s2row + k0, half);      // B = s2^T (K=d, N=t)
        acc = __builtin_amdgcn_wmma_f32_16x16x32_f16(
            false, a, false, bb, (short)0, acc, false, false);
    }
    const int t  = t0 + l15;
    const float nt = n2[b * L_ + t];
#pragma unroll
    for (int i = 0; i < 8; ++i) {
        const int s = s0 + i + 8 * half;
        float d2 = n1[b * L_ + s] + nt - 2.0f * acc[i];
        d2 = d2 > 0.0f ? d2 : 0.0f;
        const float av = 1.0f / (1.0f + sqrtf(d2));
        const _Float16 ah = (_Float16)av;
        Ah [(size_t)b * L_ * L_ + (size_t)s * L_ + t] = ah;   // A
        Aht[(size_t)b * L_ * L_ + (size_t)t * L_ + s] = ah;   // A^T
    }
}

// ---- Kernel 4: f = A @ W (z=0: A, f1; z=1: A^T, f2) via WMMA --------------
// grid (B, L/16, 2), 256 threads = 8 waves; wave w owns n-tiles 2w, 2w+1.
__global__ __launch_bounds__(256) void gemm_kernel(
    const _Float16* __restrict__ Ah, const _Float16* __restrict__ Aht,
    const _Float16* __restrict__ Wht,
    float* __restrict__ f1, float* __restrict__ f2) {
    const int b  = blockIdx.x;
    const int m0 = blockIdx.y * 16;
    const int z  = blockIdx.z;
    const _Float16* A = z ? Aht : Ah;
    float* out        = z ? f2 : f1;
    const int wave = threadIdx.x >> 5;
    const int lane = threadIdx.x & 31;
    const int half = lane >> 4;
    const int l15  = lane & 15;
    const int n0a  = (wave * 2 + 0) * 16;
    const int n0b  = (wave * 2 + 1) * 16;
    const _Float16* arow  = A   + (size_t)b * L_ * L_ + (size_t)(m0 + l15) * L_;
    const _Float16* wrowa = Wht + (size_t)(n0a + l15) * L_;
    const _Float16* wrowb = Wht + (size_t)(n0b + l15) * L_;
    v8f acc0 = {}, acc1 = {};
#pragma unroll
    for (int k0 = 0; k0 < L_; k0 += 32) {
        v16h a  = load_a_frag(arow + k0, half);       // shared A-frag
        v16h b0 = load_b_frag(wrowa + k0, half);      // W^T rows -> B frags
        v16h b1 = load_b_frag(wrowb + k0, half);
        acc0 = __builtin_amdgcn_wmma_f32_16x16x32_f16(
            false, a, false, b0, (short)0, acc0, false, false);
        acc1 = __builtin_amdgcn_wmma_f32_16x16x32_f16(
            false, a, false, b1, (short)0, acc1, false, false);
    }
#pragma unroll
    for (int i = 0; i < 8; ++i) {
        const int m = m0 + i + 8 * half;
        out[(size_t)b * L_ * D_ + (size_t)m * D_ + (n0a + l15)] = acc0[i];
        out[(size_t)b * L_ * D_ + (size_t)m * D_ + (n0b + l15)] = acc1[i];
    }
}

// ---- Kernel 5: 2-ch 3x3 conv (pad 1) + bias + tanh ------------------------
// grid (B*L*D/256, 2): y=0 -> (sent1,f1)->o1; y=1 -> (sent2,f2)->o2.
__global__ __launch_bounds__(256) void conv_tanh_kernel(
    const float* __restrict__ sent1, const float* __restrict__ sent2,
    const float* __restrict__ f1, const float* __restrict__ f2,
    const float* __restrict__ cw, const float* __restrict__ cb,
    float* __restrict__ out) {
    const int z = blockIdx.y;
    const float* xs = z ? sent2 : sent1;   // channel 0
    const float* xf = z ? f2    : f1;      // channel 1
    const int idx = blockIdx.x * 256 + threadIdx.x;  // over B*L*D
    const int d  = idx % D_;
    const int l  = (idx / D_) % L_;
    const int bb = idx / (L_ * D_);
    const float* ps = xs + (size_t)bb * L_ * D_;
    const float* pf = xf + (size_t)bb * L_ * D_;
    float acc = cb[0];
#pragma unroll
    for (int ky = 0; ky < 3; ++ky) {
        const int ly = l + ky - 1;
        if (ly < 0 || ly >= L_) continue;
#pragma unroll
        for (int kx = 0; kx < 3; ++kx) {
            const int dx = d + kx - 1;
            if (dx < 0 || dx >= D_) continue;
            acc += cw[ky * 3 + kx]     * ps[(size_t)ly * D_ + dx];
            acc += cw[9 + ky * 3 + kx] * pf[(size_t)ly * D_ + dx];
        }
    }
    out[(size_t)z * B_ * L_ * D_ + idx] = tanhf(acc);
}

extern "C" void kernel_launch(void* const* d_in, const int* in_sizes, int n_in,
                              void* d_out, int out_size, void* d_ws, size_t ws_size,
                              hipStream_t stream) {
    (void)in_sizes; (void)n_in; (void)out_size; (void)ws_size;
    const float* sent1 = (const float*)d_in[0];
    const float* sent2 = (const float*)d_in[1];
    const int*   mask1 = (const int*)d_in[2];
    const int*   mask2 = (const int*)d_in[3];
    const float* W     = (const float*)d_in[4];
    const float* cw    = (const float*)d_in[5];
    const float* cb    = (const float*)d_in[6];
    float* out = (float*)d_out;

    constexpr size_t BLD = (size_t)B_ * L_ * D_;
    constexpr size_t BLL = (size_t)B_ * L_ * L_;
    char* ws = (char*)d_ws;
    size_t off = 0;
    auto carve = [&](size_t bytes) {
        size_t o = off; off += (bytes + 255) & ~(size_t)255; return o;
    };
    _Float16* s1h = (_Float16*)(ws + carve(BLD * 2));
    _Float16* s2h = (_Float16*)(ws + carve(BLD * 2));
    float*    n1  = (float*)   (ws + carve((size_t)B_ * L_ * 4));
    float*    n2  = (float*)   (ws + carve((size_t)B_ * L_ * 4));
    _Float16* Wht = (_Float16*)(ws + carve((size_t)L_ * D_ * 2));
    _Float16* Ah  = (_Float16*)(ws + carve(BLL * 2));
    _Float16* Aht = (_Float16*)(ws + carve(BLL * 2));
    float*    f1  = (float*)   (ws + carve(BLD * 4));
    float*    f2  = (float*)   (ws + carve(BLD * 4));

    prep_rows_kernel<<<dim3(B_ * L_, 2), 256, 0, stream>>>(
        sent1, sent2, mask1, mask2, s1h, s2h, n1, n2);
    prep_w_kernel<<<dim3(L_), 256, 0, stream>>>(W, Wht);
    dist_kernel<<<dim3(B_, L_ / 16), 256, 0, stream>>>(
        s1h, s2h, n1, n2, Ah, Aht);
    gemm_kernel<<<dim3(B_, L_ / 16, 2), 256, 0, stream>>>(
        Ah, Aht, Wht, f1, f2);
    conv_tanh_kernel<<<dim3((unsigned)(BLD / 256), 2), 256, 0, stream>>>(
        sent1, sent2, f1, f2, cw, cb, out);
}
